// Network_75161927680019
// MI455X (gfx1250) — matrix operand order repaired and checked by reading, not compiled, not run on hardware
//
#include <hip/hip_runtime.h>
#include <hip/hip_bf16.h>

// ---------------------------------------------------------------------------
// CDNA5 (gfx1250) implementation of the point-transformer segmentation net.
// All linear layers run through v_wmma_f32_16x16x32_f16 (f16 in, f32 accum).
// Irregular ops (FPS / ball-query / kNN / softmax / interpolation) are wave32
// scalar kernels. Workspace layout assumes ws_size >= ~700 MB.
// ---------------------------------------------------------------------------

typedef __attribute__((ext_vector_type(16))) _Float16 v16h;
typedef __attribute__((ext_vector_type(8)))  float    v8f;

#define CEILDIV(a, b) (((a) + (b) - 1) / (b))

static constexpr int BB      = 4;
static constexpr int N_PC    = 4096;
static constexpr int N_SURF  = 9728;
static constexpr int C_IN    = 14;
static constexpr int N_CLASS = 12;

// ======================= WMMA GEMM (bias + relu + residual) =================
// out[M,N] = act( A[M,K] @ W[K,N] + bias + residual )
// Block: 128 threads = 4 waves. Tile 128(M) x 64(N), K step 32.
// Each wave computes a 32x64 strip: 2 A-fragments x 4 B-fragments = 8 WMMA/step.
#define TM 128
#define TN 64
#define TK 32

__global__ __launch_bounds__(128)
void k_gemm(const float* __restrict__ A, const float* __restrict__ W,
            const float* __restrict__ bias, const float* __restrict__ residual,
            float* __restrict__ out, int M, int K, int N, int relu) {
  __shared__ _Float16 sA[TM][TK];   // row-major A tile      (8 KB)
  __shared__ _Float16 sBT[TN][TK];  // B tile transposed     (4 KB)

  const int tid  = threadIdx.x;
  const int wave = tid >> 5;
  const int lane = tid & 31;
  const int m_blk = blockIdx.x * TM;
  const int n_blk = blockIdx.y * TN;

  v8f acc[8] = {};   // [s][nb] : s = 16-row strip (0,1), nb = 16-col block (0..3)

  for (int k0 = 0; k0 < K; k0 += TK) {
    const bool fullK = (k0 + TK <= K);
    // ---- stage A tile (f32 -> f16) ----
    if (fullK && (m_blk + TM <= M)) {
      // fast path: thread t stages one full row, fully unconditional loads
      const float* src = A + (size_t)(m_blk + tid) * K + k0;
      float v[TK];
#pragma unroll
      for (int e = 0; e < TK; ++e) v[e] = src[e];
#pragma unroll
      for (int e = 0; e < TK; ++e) sA[tid][e] = (_Float16)v[e];
    } else {
      for (int i = tid; i < TM * TK; i += 128) {
        int r = i >> 5, ccol = i & 31;
        int gm = m_blk + r, gk = k0 + ccol;
        float v = (gm < M && gk < K) ? A[(size_t)gm * K + gk] : 0.0f;
        sA[r][ccol] = (_Float16)v;
      }
    }
    // ---- stage B tile transposed ----
    if (fullK && (n_blk + TN <= N)) {
      // fast path: thread pair (n, half) stages 16 K-deep values of column n
      int n = tid >> 1, h = tid & 1;
      const float* src = W + (size_t)(k0 + h * 16) * N + n_blk + n;
      float v[16];
#pragma unroll
      for (int e = 0; e < 16; ++e) v[e] = src[(size_t)e * N];
#pragma unroll
      for (int e = 0; e < 16; ++e) sBT[n][h * 16 + e] = (_Float16)v[e];
    } else {
      for (int i = tid; i < TK * TN; i += 128) {
        int r = i >> 6, ccol = i & 63;   // r = k, ccol = n
        int gk = k0 + r, gn = n_blk + ccol;
        float v = (gk < K && gn < N) ? W[(size_t)gk * N + gn] : 0.0f;
        sBT[ccol][r] = (_Float16)v;
      }
    }
    __syncthreads();

    // ---- prefetch next K tile into cache (global_prefetch) ----
    if (k0 + TK < K) {
      int pm = m_blk + tid; pm = pm < M ? pm : M - 1;
      __builtin_prefetch(A + (size_t)pm * K + k0 + TK, 0, 1);
      if (tid < TK) {
        int wk = k0 + TK + tid; wk = wk < K ? wk : K - 1;
        __builtin_prefetch(W + (size_t)wk * N + (n_blk < N ? n_blk : N - 1), 0, 1);
      }
    }

    // ---- B fragments: 32x16 f16 each; col n = lane%16, K run of 16 ----
    const int kbb = (lane < 16) ? 0 : 16;
    const int bc  = lane & 15;
    v16h bf[4];
#pragma unroll
    for (int nb = 0; nb < 4; ++nb) {
#pragma unroll
      for (int e = 0; e < 16; ++e) bf[nb][e] = sBT[nb * 16 + bc][kbb + e];
    }

    // ---- A fragments + WMMA ----
    const int kb = (lane < 16) ? 0 : 16;
#pragma unroll
    for (int s = 0; s < 2; ++s) {
      const int arow = wave * 32 + s * 16 + (lane & 15);
      v16h af;
#pragma unroll
      for (int e = 0; e < 8; ++e) af[e]     = sA[arow][kb + e];
#pragma unroll
      for (int e = 0; e < 8; ++e) af[8 + e] = sA[arow][kb + 8 + e];
#pragma unroll
      for (int nb = 0; nb < 4; ++nb) {
        acc[s * 4 + nb] = __builtin_amdgcn_wmma_f32_16x16x32_f16(
            false, af, false, bf[nb], (short)0, acc[s * 4 + nb], false, false);
      }
    }
    __syncthreads();
  }

  // ---- store: D layout: VGPR r -> M = strip + r + (lane<16?0:8), N = nb*16 + lane%16
#pragma unroll
  for (int s = 0; s < 2; ++s) {
    const int mofs = wave * 32 + s * 16 + ((lane < 16) ? 0 : 8);
#pragma unroll
    for (int nb = 0; nb < 4; ++nb) {
      v8f a = acc[s * 4 + nb];
      int gn = n_blk + nb * 16 + (lane & 15);
      if (gn >= N) continue;
      float bv = bias ? bias[gn] : 0.0f;
#pragma unroll
      for (int r = 0; r < 8; ++r) {
        int gm = m_blk + mofs + r;
        if (gm < M) {
          float v = a[r] + bv;
          if (residual) v += residual[(size_t)gm * N + gn];
          if (relu) v = fmaxf(v, 0.0f);
          out[(size_t)gm * N + gn] = v;
        }
      }
    }
  }
}

// ============================ utility kernels ==============================

__global__ void k_split(const float* __restrict__ in, float* __restrict__ xyz,
                        float* __restrict__ feat, int N, int C) {
  int t = blockIdx.x * blockDim.x + threadIdx.x;
  int tot = BB * N;
  if (t >= tot) return;
  const float* row = in + (size_t)t * (3 + C);
  float* x = xyz + (size_t)t * 3;
  x[0] = row[0]; x[1] = row[1]; x[2] = row[2];
  float* f = feat + (size_t)t * C;
  for (int c = 0; c < C; ++c) f[c] = row[3 + c];
}

__global__ void k_labels(const float* __restrict__ gt, int* __restrict__ lab, int tot) {
  int t = blockIdx.x * blockDim.x + threadIdx.x;
  if (t >= tot) return;
  const float* g = gt + (size_t)t * N_CLASS;
  int best = 0; float bv = g[0];
  for (int c = 1; c < N_CLASS; ++c) if (g[c] > bv) { bv = g[c]; best = c; }
  lab[t] = best;
}

// Iterative farthest point sampling: one block per batch element.
__global__ __launch_bounds__(256)
void k_fps(const float* __restrict__ xyz, int N, int npoint,
           int* __restrict__ fidx, float* __restrict__ distbuf) {
  int b = blockIdx.x;
  const float* x = xyz + (size_t)b * N * 3;
  float* dist = distbuf + (size_t)b * N;
  __shared__ float sval[256];
  __shared__ int   sidx[256];
  int tid = threadIdx.x;
  for (int i = tid; i < N; i += 256) dist[i] = 1e10f;
  __syncthreads();
  int far = 0;
  for (int it = 0; it < npoint; ++it) {
    if (tid == 0) fidx[(size_t)b * npoint + it] = far;
    float cx = x[3 * far], cy = x[3 * far + 1], cz = x[3 * far + 2];
    float best = -1.0f; int bi = 0;
    for (int i = tid; i < N; i += 256) {
      float dx = x[3 * i] - cx, dy = x[3 * i + 1] - cy, dz = x[3 * i + 2] - cz;
      float d = dx * dx + dy * dy + dz * dz;
      float m = fminf(dist[i], d);
      dist[i] = m;
      if (m > best) { best = m; bi = i; }
    }
    sval[tid] = best; sidx[tid] = bi;
    __syncthreads();
    for (int s = 128; s > 0; s >>= 1) {
      if (tid < s && sval[tid + s] > sval[tid]) { sval[tid] = sval[tid + s]; sidx[tid] = sidx[tid + s]; }
      __syncthreads();
    }
    far = sidx[0];
    __syncthreads();
  }
}

__global__ void k_gather_rows(const float* __restrict__ src, const int* __restrict__ idx,
                              float* __restrict__ out, int nIdx, int srcN, int C) {
  long t = (long)blockIdx.x * blockDim.x + threadIdx.x;
  long tot = (long)BB * nIdx * C;
  if (t >= tot) return;
  int c = (int)(t % C);
  int i = (int)((t / C) % nIdx);
  int b = (int)(t / ((long)C * nIdx));
  int id = idx[(size_t)b * nIdx + i];
  out[t] = src[((size_t)b * srcN + id) * C + c];
}

__global__ void k_gather_int(const int* __restrict__ src, const int* __restrict__ idx,
                             int* __restrict__ out, int nIdx, int srcN) {
  int t = blockIdx.x * blockDim.x + threadIdx.x;
  int tot = BB * nIdx;
  if (t >= tot) return;
  int b = t / nIdx, i = t % nIdx;
  out[t] = src[(size_t)b * srcN + idx[(size_t)b * nIdx + i]];
}

// Ball query + group + normalize + concat feat -> [B, npoint, nsample, 3+C]
__global__ void k_ball_group(const float* __restrict__ xyz, const float* __restrict__ feat,
                             const float* __restrict__ new_xyz, int N, int npoint,
                             int nsample, int C, float radius, float* __restrict__ out) {
  int b = blockIdx.y;
  int j = blockIdx.x * blockDim.x + threadIdx.x;
  if (j >= npoint) return;
  const float* x = xyz + (size_t)b * N * 3;
  const float* f = feat + (size_t)b * N * C;
  const float* q = new_xyz + ((size_t)b * npoint + j) * 3;
  float qx = q[0], qy = q[1], qz = q[2];
  float r2 = radius * radius, inv = 1.0f / radius;
  int Cg = 3 + C;
  float* o = out + ((size_t)b * npoint + j) * nsample * Cg;
  int cnt = 0;
  for (int i = 0; i < N && cnt < nsample; ++i) {
    float dx = x[3 * i] - qx, dy = x[3 * i + 1] - qy, dz = x[3 * i + 2] - qz;
    if (dx * dx + dy * dy + dz * dz < r2) {
      float* row = o + (size_t)cnt * Cg;
      row[0] = dx * inv; row[1] = dy * inv; row[2] = dz * inv;
      for (int c = 0; c < C; ++c) row[3 + c] = f[(size_t)i * C + c];
      ++cnt;
    }
  }
  if (cnt == 0) {  // reference maps empty balls to index 0
    float dx = x[0] - qx, dy = x[1] - qy, dz = x[2] - qz;
    o[0] = dx * inv; o[1] = dy * inv; o[2] = dz * inv;
    for (int c = 0; c < C; ++c) o[3 + c] = f[c];
    cnt = 1;
  }
  for (; cnt < nsample; ++cnt) {
    float* row = o + (size_t)cnt * Cg;
    for (int c = 0; c < Cg; ++c) row[c] = o[c];
  }
}

__global__ void k_maxpool(const float* __restrict__ in, float* __restrict__ out,
                          int rows, int ns, int C) {
  long t = (long)blockIdx.x * blockDim.x + threadIdx.x;
  long tot = (long)rows * C;
  if (t >= tot) return;
  int c = (int)(t % C);
  long r = t / C;
  const float* p = in + (r * ns) * (size_t)C + c;
  float m = p[0];
  for (int s = 1; s < ns; ++s) m = fmaxf(m, p[(size_t)s * C]);
  out[t] = m;
}

template <int KK>
__global__ void k_knn(const float* __restrict__ qxyz, const float* __restrict__ kxyz,
                      int Nq, int Nkv, int* __restrict__ knn) {
  int b = blockIdx.y;
  int n = blockIdx.x * blockDim.x + threadIdx.x;
  if (n >= Nq) return;
  const float* q = qxyz + ((size_t)b * Nq + n) * 3;
  const float* x = kxyz + (size_t)b * Nkv * 3;
  float qx = q[0], qy = q[1], qz = q[2];
  float bd[KK]; int bi[KK];
#pragma unroll
  for (int t = 0; t < KK; ++t) { bd[t] = 3.4e38f; bi[t] = 0; }
  for (int i = 0; i < Nkv; ++i) {
    float dx = x[3 * i] - qx, dy = x[3 * i + 1] - qy, dz = x[3 * i + 2] - qz;
    float d = dx * dx + dy * dy + dz * dz;
    if (d < bd[KK - 1]) {
      int p = KK - 1;
      while (p > 0 && bd[p - 1] > d) { bd[p] = bd[p - 1]; bi[p] = bi[p - 1]; --p; }
      bd[p] = d; bi[p] = i;
    }
  }
  int* o = knn + ((size_t)b * Nq + n) * KK;
  for (int t = 0; t < KK; ++t) o[t] = bi[t];
}

__global__ void k_rel(const float* __restrict__ qxyz, const float* __restrict__ kvxyz,
                      const int* __restrict__ knn, float* __restrict__ rel,
                      int Nq, int Nkv, int KK) {
  long t = (long)blockIdx.x * blockDim.x + threadIdx.x;
  long tot = (long)BB * Nq * KK;
  if (t >= tot) return;
  int n = (int)((t / KK) % Nq);
  int b = (int)(t / ((long)KK * Nq));
  int id = knn[t];
  const float* q = qxyz + ((size_t)b * Nq + n) * 3;
  const float* p = kvxyz + ((size_t)b * Nkv + id) * 3;
  float* o = rel + (size_t)t * 3;
  o[0] = q[0] - p[0]; o[1] = q[1] - p[1]; o[2] = q[2] - p[2];
}

// h = q (broadcast over k) - kf + pos
__global__ void k_h(const float* __restrict__ q, const float* __restrict__ kf,
                    const float* __restrict__ pos, float* __restrict__ h,
                    int Nq, int KK, int D) {
  long t = (long)blockIdx.x * blockDim.x + threadIdx.x;
  long tot = (long)BB * Nq * KK * D;
  if (t >= tot) return;
  int d = (int)(t % D);
  long bn = t / ((long)D * KK);   // b*Nq + n
  h[t] = q[bn * D + d] - kf[t] + pos[t];
}

__global__ void k_add(const float* __restrict__ a, const float* __restrict__ b,
                      float* __restrict__ o, long n) {
  long t = (long)blockIdx.x * blockDim.x + threadIdx.x;
  if (t < n) o[t] = a[t] + b[t];
}

// softmax over k (with optional label mask) fused with weighted sum of vp.
template <int KK>
__global__ void k_softmax_wsum(const float* __restrict__ logits, const float* __restrict__ vp,
                               const int* __restrict__ knn, const int* __restrict__ lab,
                               float scale, float* __restrict__ out, int Nq, int D) {
  long t = (long)blockIdx.x * blockDim.x + threadIdx.x;
  long tot = (long)BB * Nq * D;
  if (t >= tot) return;
  int d = (int)(t % D);
  int n = (int)((t / D) % Nq);
  int b = (int)(t / ((long)D * Nq));
  size_t row = (size_t)b * Nq + n;
  const float* lg = logits + row * KK * (size_t)D + d;
  const float* vv = vp     + row * KK * (size_t)D + d;
  int myl = lab ? lab[row] : 0;
  float e[KK]; float mx = -3.4e38f;
#pragma unroll
  for (int i = 0; i < KK; ++i) {
    float L = lg[(size_t)i * D] * scale;
    if (lab) {
      int kn = knn[row * KK + i];
      if (lab[(size_t)b * Nq + kn] != myl) L = -1e9f;
    }
    e[i] = L; mx = fmaxf(mx, L);
  }
  float s = 0.0f;
#pragma unroll
  for (int i = 0; i < KK; ++i) { e[i] = __expf(e[i] - mx); s += e[i]; }
  float invs = 1.0f / s, acc = 0.0f;
#pragma unroll
  for (int i = 0; i < KK; ++i) acc += e[i] * invs * vv[(size_t)i * D];
  out[row * D + d] = acc;
}

// 3-NN inverse-distance interpolation + concat(feat1, interp(feat2))
__global__ void k_interp_concat(const float* __restrict__ xyz1, const float* __restrict__ xyz2,
                                const float* __restrict__ f1, const float* __restrict__ f2,
                                float* __restrict__ out, int N1, int N2, int C1, int C2) {
  int t = blockIdx.x * blockDim.x + threadIdx.x;
  int tot = BB * N1;
  if (t >= tot) return;
  int b = t / N1, n = t % N1;
  const float* q  = xyz1 + ((size_t)b * N1 + n) * 3;
  const float* x2 = xyz2 + (size_t)b * N2 * 3;
  float qx = q[0], qy = q[1], qz = q[2];
  float bd0 = 3.4e38f, bd1 = 3.4e38f, bd2 = 3.4e38f;
  int bi0 = 0, bi1 = 0, bi2 = 0;
  for (int i = 0; i < N2; ++i) {
    float dx = x2[3 * i] - qx, dy = x2[3 * i + 1] - qy, dz = x2[3 * i + 2] - qz;
    float d = dx * dx + dy * dy + dz * dz;
    if (d < bd2) {
      if (d < bd0)      { bd2 = bd1; bi2 = bi1; bd1 = bd0; bi1 = bi0; bd0 = d; bi0 = i; }
      else if (d < bd1) { bd2 = bd1; bi2 = bi1; bd1 = d; bi1 = i; }
      else              { bd2 = d; bi2 = i; }
    }
  }
  float w0 = 1.0f / (bd0 + 1e-8f), w1 = 1.0f / (bd1 + 1e-8f), w2 = 1.0f / (bd2 + 1e-8f);
  float ws = w0 + w1 + w2; w0 /= ws; w1 /= ws; w2 /= ws;
  float* o = out + ((size_t)b * N1 + n) * (C1 + C2);
  const float* r1 = f1 + ((size_t)b * N1 + n) * C1;
  for (int c = 0; c < C1; ++c) o[c] = r1[c];
  const float* p0 = f2 + ((size_t)b * N2 + bi0) * C2;
  const float* p1 = f2 + ((size_t)b * N2 + bi1) * C2;
  const float* p2 = f2 + ((size_t)b * N2 + bi2) * C2;
  for (int c = 0; c < C2; ++c) o[C1 + c] = w0 * p0[c] + w1 * p1[c] + w2 * p2[c];
}

__global__ void k_transpose_out(const float* __restrict__ in, float* __restrict__ out,
                                int N, int C) {
  long t = (long)blockIdx.x * blockDim.x + threadIdx.x;
  long tot = (long)BB * N * C;
  if (t >= tot) return;
  int c = (int)(t % C);
  int n = (int)((t / C) % N);
  int b = (int)(t / ((long)C * N));
  out[((size_t)b * C + c) * N + n] = in[t];
}

// =============================== host side ==================================

struct Lin { const float* w; const float* b; };
struct TFp { Lin d0, d1, fc1, fc2, g0, g1, mh; const float *wk, *wq, *wv; };

struct Ctx {
  float *G0, *G1;                          // big arenas (grouped tensors)
  float *KF, *VF, *POS, *H;                // attention temporaries (carved from G0)
  float *Xq, *Xkv, *Q, *KA, *VA, *RES, *REL;
  float *CC, *FT;
  float *dist;
  int   *knn;
  hipStream_t s;
};

static void gemm(Ctx& c, const float* A, int M, int K, const float* W,
                 const float* bias, int N, bool relu, const float* residual, float* out) {
  dim3 g(CEILDIV(M, TM), CEILDIV(N, TN));
  k_gemm<<<g, 128, 0, c.s>>>(A, W, bias, residual, out, M, K, N, relu ? 1 : 0);
}

static void sa_module(Ctx& c, const Lin* mlp, const int* dims,
                      const float* xyz, const float* feat, int N, int C,
                      int npoint, float radius, int nsample,
                      float* new_xyz, float* new_feat, int* fidx) {
  k_fps<<<BB, 256, 0, c.s>>>(xyz, N, npoint, fidx, c.dist);
  k_gather_rows<<<CEILDIV(BB * npoint * 3, 256), 256, 0, c.s>>>(xyz, fidx, new_xyz, npoint, N, 3);
  dim3 gb(CEILDIV(npoint, 128), BB);
  k_ball_group<<<gb, 128, 0, c.s>>>(xyz, feat, new_xyz, N, npoint, nsample, C, radius, c.G0);
  int M = BB * npoint * nsample;
  gemm(c, c.G0, M, 3 + C,   mlp[0].w, mlp[0].b, dims[0], true, nullptr, c.G1);
  gemm(c, c.G1, M, dims[0], mlp[1].w, mlp[1].b, dims[1], true, nullptr, c.G0);
  gemm(c, c.G0, M, dims[1], mlp[2].w, mlp[2].b, dims[2], true, nullptr, c.G1);
  k_maxpool<<<CEILDIV(BB * npoint * dims[2], 256), 256, 0, c.s>>>(c.G1, new_feat, BB * npoint, nsample, dims[2]);
}

static void tf_block(Ctx& c, const TFp& p, int KK,
                     const float* xyz_q, const float* feat_q, int Nq, int Cq,
                     const float* xyz_kv, const float* feat_kv, int Nkv, int Ckv,
                     const int* lab, float* out) {
  bool self = (xyz_kv == nullptr);
  if (self) { xyz_kv = xyz_q; feat_kv = feat_q; Nkv = Nq; Ckv = Cq; }
  dim3 gk(CEILDIV(Nq, 128), BB);
  if (KK == 16) k_knn<16><<<gk, 128, 0, c.s>>>(xyz_q, xyz_kv, Nq, Nkv, c.knn);
  else          k_knn<32><<<gk, 128, 0, c.s>>>(xyz_q, xyz_kv, Nq, Nkv, c.knn);

  const int Mq = BB * Nq, Mkv = BB * Nkv;
  const long MK = (long)Mq * KK;

  gemm(c, feat_q, Mq, Cq, p.fc1.w, p.fc1.b, 64, false, nullptr, c.Xq);
  const float* Xkv = c.Xq;
  if (!self) { gemm(c, feat_kv, Mkv, Ckv, p.fc1.w, p.fc1.b, 64, false, nullptr, c.Xkv); Xkv = c.Xkv; }
  gemm(c, c.Xq, Mq, 64, p.wq, nullptr, 64, false, nullptr, c.Q);
  gemm(c, Xkv, Mkv, 64, p.wk, nullptr, 64, false, nullptr, c.KA);
  gemm(c, Xkv, Mkv, 64, p.wv, nullptr, 64, false, nullptr, c.VA);

  k_gather_rows<<<CEILDIV(MK * 64, 256), 256, 0, c.s>>>(c.KA, c.knn, c.KF, Nq * KK, Nkv, 64);
  k_gather_rows<<<CEILDIV(MK * 64, 256), 256, 0, c.s>>>(c.VA, c.knn, c.VF, Nq * KK, Nkv, 64);
  k_rel<<<CEILDIV(MK, 256), 256, 0, c.s>>>(xyz_q, xyz_kv, c.knn, c.REL, Nq, Nkv, KK);

  // pos = delta MLP (3 -> 64 relu -> 64)
  gemm(c, c.REL, (int)MK, 3,  p.d0.w, p.d0.b, 64, true,  nullptr, c.H);
  gemm(c, c.H,   (int)MK, 64, p.d1.w, p.d1.b, 64, false, nullptr, c.POS);
  // h = q - kf + pos
  k_h<<<CEILDIV(MK * 64, 256), 256, 0, c.s>>>(c.Q, c.KF, c.POS, c.H, Nq, KK, 64);
  // vp = vf + pos (in place into VF)
  k_add<<<CEILDIV(MK * 64, 256), 256, 0, c.s>>>(c.VF, c.POS, c.VF, MK * 64);
  // logits = gamma MLP on h (64 -> 64 relu -> 64); reuse POS and KF
  gemm(c, c.H,   (int)MK, 64, p.g0.w, p.g0.b, 64, true,  nullptr, c.POS);
  gemm(c, c.POS, (int)MK, 64, p.g1.w, p.g1.b, 64, false, nullptr, c.KF);

  const float scale = 0.125f;  // 1/sqrt(64)
  long totD = (long)Mq * 64;
  if (KK == 16)
    k_softmax_wsum<16><<<CEILDIV(totD, 256), 256, 0, c.s>>>(c.KF, c.VF, c.knn, lab, scale, c.RES, Nq, 64);
  else
    k_softmax_wsum<32><<<CEILDIV(totD, 256), 256, 0, c.s>>>(c.KF, c.VF, c.knn, lab, scale, c.RES, Nq, 64);

  gemm(c, c.RES, Mq, 64, p.fc2.w, p.fc2.b, Cq, false, feat_q, out);
}

static void fp_module(Ctx& c, const Lin* mlp, const int* dims,
                      const float* xyz1, int N1, const float* xyz2, int N2,
                      const float* f1, int C1, const float* f2, int C2, float* out) {
  k_interp_concat<<<CEILDIV(BB * N1, 128), 128, 0, c.s>>>(xyz1, xyz2, f1, f2, c.CC, N1, N2, C1, C2);
  int M = BB * N1;
  gemm(c, c.CC, M, C1 + C2, mlp[0].w, mlp[0].b, dims[0], true, nullptr, c.FT);
  gemm(c, c.FT, M, dims[0], mlp[1].w, mlp[1].b, dims[1], true, nullptr, out);
}

extern "C" void kernel_launch(void* const* d_in, const int* in_sizes, int n_in,
                              void* d_out, int out_size, void* d_ws, size_t ws_size,
                              hipStream_t stream) {
  (void)in_sizes; (void)n_in; (void)out_size; (void)ws_size;

  const float* pc = (const float*)d_in[0];
  const float* ps = (const float*)d_in[1];
  const float* gt = (const float*)d_in[2];

  // ---- parameter walk: jax tree-flatten (sorted dict keys, 'b' before 'w') ----
  int ii = 3;
  auto takeW   = [&]() { return (const float*)d_in[ii++]; };
  auto takeLin = [&]() { Lin l; l.b = takeW(); l.w = takeW(); return l; };
  auto takeTF  = [&](bool masked) {
    TFp t;
    t.d0 = takeLin(); t.d1 = takeLin();
    t.fc1 = takeLin(); t.fc2 = takeLin();
    t.g0 = takeLin(); t.g1 = takeLin();
    if (masked) t.mh = takeLin(); else { t.mh.w = nullptr; t.mh.b = nullptr; }
    t.wk = takeW(); t.wq = takeW(); t.wv = takeW();
    return t;
  };
  TFp atten1  = takeTF(false);
  TFp atten1s = takeTF(false);
  TFp atten2  = takeTF(false);
  TFp atten3  = takeTF(false);
  TFp atten4  = takeTF(false);
  TFp atten5  = takeTF(true);
  TFp atten6  = takeTF(true);
  TFp atten7  = takeTF(true);
  Lin dimL    = takeLin();
  Lin fp1[2]  = { takeLin(), takeLin() };
  Lin fp2[2]  = { takeLin(), takeLin() };
  Lin fp3[2]  = { takeLin(), takeLin() };
  Lin fp4[2]  = { takeLin(), takeLin() };
  Lin sa1[3]  = { takeLin(), takeLin(), takeLin() };
  Lin sa1s[3] = { takeLin(), takeLin(), takeLin() };
  Lin sa2[3]  = { takeLin(), takeLin(), takeLin() };
  Lin sa2s[3] = { takeLin(), takeLin(), takeLin() };
  Lin sa3[3]  = { takeLin(), takeLin(), takeLin() };
  Lin sa4[3]  = { takeLin(), takeLin(), takeLin() };

  // ---- workspace bump allocator ----
  size_t off = 0;
  auto allocF = [&](size_t count) {
    float* p = (float*)((char*)d_ws + off);
    off = (off + count * sizeof(float) + 255) & ~(size_t)255;
    return p;
  };
  auto allocI = [&](size_t count) {
    int* p = (int*)((char*)d_ws + off);
    off = (off + count * sizeof(int) + 255) & ~(size_t)255;
    return p;
  };

  Ctx c; c.s = stream;
  const size_t ARENA = 67200000;   // >= 4*2048*64*128 floats
  const size_t ATTNB = 16800000;   // >= 4*2048*32*64 floats
  c.G0  = allocF(ARENA);
  c.G1  = allocF(ARENA);
  c.KF  = c.G0;                    // attention temporaries alias arena 0
  c.VF  = c.G0 + ATTNB;
  c.POS = c.G0 + 2 * ATTNB;
  c.H   = c.G0 + 3 * ATTNB;

  c.dist = allocF((size_t)BB * N_SURF);
  c.knn  = allocI((size_t)BB * 2048 * 32);
  c.Xq   = allocF((size_t)BB * 2048 * 64);
  c.Xkv  = allocF((size_t)BB * 2048 * 64);
  c.Q    = allocF((size_t)BB * 2048 * 64);
  c.KA   = allocF((size_t)BB * 2048 * 64);
  c.VA   = allocF((size_t)BB * 2048 * 64);
  c.RES  = allocF((size_t)BB * 2048 * 64);
  c.REL  = allocF((size_t)BB * 2048 * 32 * 3);
  c.CC   = allocF((size_t)BB * 4096 * 512);
  c.FT   = allocF((size_t)BB * 4096 * 256);

  int* iS1 = allocI(BB * 512);
  int* iS2 = allocI(BB * 256);
  int* i1  = allocI(BB * 2048);
  int* i2  = allocI(BB * 1024);
  int* i3  = allocI(BB * 512);
  int* i4  = allocI(BB * 256);
  int* lab  = allocI((size_t)BB * N_PC);
  int* lab1 = allocI(BB * 2048);
  int* lab2 = allocI(BB * 1024);
  int* lab3 = allocI(BB * 512);

  float* xyz0   = allocF((size_t)BB * N_PC * 3);
  float* feat0  = allocF((size_t)BB * N_PC * C_IN);
  float* xyzS0  = allocF((size_t)BB * N_SURF * 3);
  float* featS0 = allocF((size_t)BB * N_SURF * C_IN);
  float* xyzS1  = allocF(BB * 512 * 3);
  float* featS1 = allocF(BB * 512 * 128);
  float* xyzS2  = allocF(BB * 256 * 3);
  float* featS2 = allocF(BB * 256 * 128);
  float* xyz1   = allocF(BB * 2048 * 3);
  float* f1a    = allocF((size_t)BB * 2048 * 128);
  float* f1b    = allocF((size_t)BB * 2048 * 128);
  float* feat1  = allocF((size_t)BB * 2048 * 128);
  float* xyz2   = allocF(BB * 1024 * 3);
  float* f2a    = allocF((size_t)BB * 1024 * 256);
  float* feat2  = allocF((size_t)BB * 1024 * 256);
  float* xyz3   = allocF(BB * 512 * 3);
  float* f3a    = allocF((size_t)BB * 512 * 256);
  float* feat3  = allocF((size_t)BB * 512 * 256);
  float* xyz4   = allocF(BB * 256 * 3);
  float* f4a    = allocF((size_t)BB * 256 * 256);
  float* feat4  = allocF((size_t)BB * 256 * 256);
  float* fA     = allocF((size_t)BB * 4096 * 256);
  float* fB     = allocF((size_t)BB * 4096 * 256);
  float* dimtmp = allocF((size_t)BB * 4096 * N_CLASS);

  // ---- forward pass ----
  k_split<<<CEILDIV(BB * N_PC, 256), 256, 0, stream>>>(pc, xyz0, feat0, N_PC, C_IN);
  k_split<<<CEILDIV(BB * N_SURF, 256), 256, 0, stream>>>(ps, xyzS0, featS0, N_SURF, C_IN);
  k_labels<<<CEILDIV(BB * N_PC, 256), 256, 0, stream>>>(gt, lab, BB * N_PC);

  const int dA[3] = {64, 64, 128};
  const int dS2[3] = {256, 256, 128};
  const int dB[3] = {128, 128, 256};
  const int d256[2] = {256, 256};

  // surface branch
  sa_module(c, sa1s, dA,  xyzS0, featS0, N_SURF, C_IN, 512, 0.2f, 64, xyzS1, featS1, iS1);
  sa_module(c, sa2s, dS2, xyzS1, featS1, 512, 128,     256, 0.4f, 32, xyzS2, featS2, iS2);

  // main branch
  sa_module(c, sa1, dA, xyz0, feat0, N_PC, C_IN, 2048, 0.2f, 64, xyz1, f1a, i1);
  tf_block(c, atten1s, 32, xyz1, f1a, 2048, 128, xyzS2, featS2, 256, 128, nullptr, f1b);
  tf_block(c, atten1, 32, xyz1, f1b, 2048, 128, nullptr, nullptr, 0, 0, nullptr, feat1);
  k_gather_int<<<CEILDIV(BB * 2048, 256), 256, 0, stream>>>(lab, i1, lab1, 2048, N_PC);

  sa_module(c, sa2, dB, xyz1, feat1, 2048, 128, 1024, 0.4f, 32, xyz2, f2a, i2);
  tf_block(c, atten2, 16, xyz2, f2a, 1024, 256, nullptr, nullptr, 0, 0, nullptr, feat2);
  k_gather_int<<<CEILDIV(BB * 1024, 256), 256, 0, stream>>>(lab, i2, lab2, 1024, N_PC);

  sa_module(c, sa3, dB, xyz2, feat2, 1024, 256, 512, 0.8f, 16, xyz3, f3a, i3);
  tf_block(c, atten3, 16, xyz3, f3a, 512, 256, nullptr, nullptr, 0, 0, nullptr, feat3);
  k_gather_int<<<CEILDIV(BB * 512, 256), 256, 0, stream>>>(lab, i3, lab3, 512, N_PC);

  sa_module(c, sa4, dB, xyz3, feat3, 512, 256, 256, 1.2f, 16, xyz4, f4a, i4);
  tf_block(c, atten4, 16, xyz4, f4a, 256, 256, nullptr, nullptr, 0, 0, nullptr, feat4);

  // decoder
  fp_module(c, fp1, d256, xyz3, 512, xyz4, 256, feat3, 256, feat4, 256, fA);
  tf_block(c, atten5, 16, xyz3, fA, 512, 256, nullptr, nullptr, 0, 0, lab3, fB);
  fp_module(c, fp2, d256, xyz2, 1024, xyz3, 512, feat2, 256, fB, 256, fA);
  tf_block(c, atten6, 16, xyz2, fA, 1024, 256, nullptr, nullptr, 0, 0, lab2, fB);
  fp_module(c, fp3, d256, xyz1, 2048, xyz2, 1024, feat1, 128, fB, 256, fA);
  tf_block(c, atten7, 32, xyz1, fA, 2048, 256, nullptr, nullptr, 0, 0, lab1, fB);
  fp_module(c, fp4, d256, xyz0, N_PC, xyz1, 2048, feat0, C_IN, fB, 256, fA);

  gemm(c, fA, BB * N_PC, 256, dimL.w, dimL.b, N_CLASS, false, nullptr, dimtmp);
  k_transpose_out<<<CEILDIV((long)BB * N_PC * N_CLASS, 256), 256, 0, stream>>>(
      dimtmp, (float*)d_out, N_PC, N_CLASS);
}
